// Select_Layer_68350109549101
// MI455X (gfx1250) — matrix operand order repaired and checked by reading, not compiled, not run on hardware
//
#include <hip/hip_runtime.h>

#define DK      256   // feature dim (fixed by problem)
#define TOPK    50
#define NSPLIT  16
#define BM      32    // batch rows per block
#define BN      64    // score columns per chunk
#define LDSTR   272   // bf16 elements per staged LDS row (256 + 16 pad vs bank conflicts)
#define NEG_INF (-3.4028235e38f)

#define SB_BYTES (2 * BN * LDSTR * 2)   // double-buffered B tile      (69632)
#define SS_BYTES (BM * BN * 4)          // score tile                  (8192)
#define TS_BYTES (BM * TOPK * 4)        // top-k scores / indices      (6400 each)
#define SMEM_TOTAL (SB_BYTES + SS_BYTES + 2 * TS_BYTES + 2 * BM * 4)

typedef __attribute__((ext_vector_type(16))) __bf16 v16bf;
typedef __attribute__((ext_vector_type(8)))  float  v8f;

union VecBF {
    v16bf v;
    uint4 q[2];
    unsigned short h[16];
};

__device__ __forceinline__ unsigned short f2u(float f) {
    union { float f; unsigned u; } x; x.f = f;
    unsigned r = x.u + 0x7FFFu + ((x.u >> 16) & 1u);   // round-to-nearest-even
    return (unsigned short)(r >> 16);
}

#if __has_builtin(__builtin_amdgcn_tensor_load_to_lds)
#define USE_TDM 1
typedef __attribute__((ext_vector_type(4))) unsigned int v4u;
typedef __attribute__((ext_vector_type(8))) int v8i;
typedef __attribute__((ext_vector_type(4))) int v4i;

// Stage a [tile1 x 256] bf16 tile (contiguous rows) from global into LDS via TDM.
// Descriptor per CDNA5 ISA: 2D tensor, data_size=2B,
// LDS padding of 8 DWORDs every 128 DWORDs (row = 512B) -> LDSTR=272 elements.
// This toolchain exposes the 6-arg builtin: (v4u, v8i, v4i, v4i, v8i, i32 cpol).
__device__ __forceinline__ void tdm_stage(const unsigned short* gsrc, unsigned ldsAddr,
                                          int dim1, int tile1) {
    unsigned long long ga = (unsigned long long)(uintptr_t)gsrc;
    v4u g0;
    g0[0] = 1u;                                              // count=1 valid descriptor
    g0[1] = ldsAddr;                                         // lds_addr [63:32]
    g0[2] = (unsigned)(ga & 0xFFFFFFFFull);                  // global_addr low
    g0[3] = (unsigned)((ga >> 32) & 0x01FFFFFFull)           // global_addr high [120:96]
          | (2u << 30);                                      // type=2 ("image")
    v8i g1;
    g1[0] = (int)((1u << 16)       // data_size = 1 -> 2 bytes
                | (1u << 20)       // pad_enable
                | (6u << 22)       // pad_interval code 6 = 128 DWORDs (one 512B row)
                | (7u << 25));     // pad_amount code 7 = 8 DWORDs (32B)
    g1[1] = (int)(256u << 16);                               // tensor_dim0[15:0] = 256
    g1[2] = (int)((((unsigned)dim1) & 0xFFFFu) << 16);       // dim0 hi=0 | tensor_dim1 lo
    g1[3] = (int)(((((unsigned)dim1) >> 16) & 0xFFFFu)       // tensor_dim1 hi
                | (256u << 16));                             // tile_dim0 = 256
    g1[4] = tile1 & 0xFFFF;                                  // tile_dim1 ; tile_dim2=0
    g1[5] = 256;                                             // tensor_dim0_stride low32
    g1[6] = 0;
    g1[7] = 0;
    v4i g2 = {0, 0, 0, 0};
    v4i g3 = {0, 0, 0, 0};
    v8i g4 = {0, 0, 0, 0, 0, 0, 0, 0};
    __builtin_amdgcn_tensor_load_to_lds(g0, g1, g2, g3, g4, 0);
}
#else
#define USE_TDM 0
#endif

// ---------------- Kernel 1: fp32 -> bf16 convert (8 elements/thread/iter) ----
__global__ void cvt_kernel(const float* __restrict__ in,
                           unsigned short* __restrict__ ob, long long n8) {
    long long i = (long long)blockIdx.x * blockDim.x + threadIdx.x;
    long long stride = (long long)gridDim.x * blockDim.x;
    for (; i < n8; i += stride) {
        const float4* p = (const float4*)(in + i * 8);
        float4 a = p[0], b = p[1];
        union { uint4 q; unsigned short h[8]; } u;
        u.h[0] = f2u(a.x); u.h[1] = f2u(a.y); u.h[2] = f2u(a.z); u.h[3] = f2u(a.w);
        u.h[4] = f2u(b.x); u.h[5] = f2u(b.y); u.h[6] = f2u(b.z); u.h[7] = f2u(b.w);
        *((uint4*)(ob + i * 8)) = u.q;
    }
}

// ---------------- Kernel 2: bs_feature = gather(idx) @ W.T + b  (WMMA bf16) --
__global__ void bs_kernel(const int* __restrict__ idx,
                          const float* __restrict__ idf,
                          const float* __restrict__ W,
                          const float* __restrict__ bias,
                          unsigned short* __restrict__ bsb, int B, int N) {
    int lane = threadIdx.x & 31;
    int jt = blockIdx.x;           // column tile (output feature j)
    int bt = blockIdx.y;           // batch row block
    int m = (bt << 4) + (lane & 15);
    int mr = m < B ? m : B - 1;
    int ur = idx[mr];
    if (ur < 0) ur = 0; if (ur >= N) ur = N - 1;
    const float* xr = idf + (size_t)ur * DK;
    int jn = (jt << 4) + (lane & 15);
    const float* wr = W + (size_t)jn * DK;     // row j of W is contiguous in K
    int ha = (lane >= 16) ? 16 : 0;
    int hb = (lane >= 16) ? 16 : 0;
    v8f acc = {};
#pragma unroll
    for (int kk = 0; kk < 8; ++kk) {
        VecBF a, bf;
        int d0 = kk * 32 + ha;
#pragma unroll
        for (int t = 0; t < 8; ++t) {
            a.h[t]     = f2u(xr[d0 + t]);
            a.h[8 + t] = f2u(xr[d0 + 8 + t]);
        }
        int db = kk * 32 + hb;
#pragma unroll
        for (int t = 0; t < 16; ++t) bf.h[t] = f2u(wr[db + t]);
        acc = __builtin_amdgcn_wmma_f32_16x16x32_bf16(false, a.v, false, bf.v,
                                                      (short)0, acc, false, false);
    }
    int n = lane & 15, mb = (lane >= 16) ? 8 : 0;
#pragma unroll
    for (int v = 0; v < 8; ++v) {
        int row = (bt << 4) + v + mb;
        int j = (jt << 4) + n;
        bsb[(size_t)row * DK + j] = f2u(acc[v] + bias[j]);
    }
}

// ---------------- Kernel 3: fused score GEMM + running top-50 ----------------
// grid = (NSPLIT, B/BM), block = 256 (8 waves). Each block: 32 rows x N/NSPLIT cols.
// B tiles double-buffered in LDS via TDM; wave w computes 16x16 tile
// (row half = w>>2, col tile = w&3).
__global__ void __launch_bounds__(256)
score_topk_kernel(const unsigned short* __restrict__ bsb,
                  const unsigned short* __restrict__ idb,
                  float* __restrict__ pS, int* __restrict__ pI, int N) {
    extern __shared__ unsigned char smem[];
    unsigned short* sB   = (unsigned short*)smem;                       // [2][BN*LDSTR]
    float*          sS   = (float*)(smem + SB_BYTES);                   // [BM][BN]
    float*          sTopS = (float*)(smem + SB_BYTES + SS_BYTES);       // [BM][TOPK]
    int*            sTopI = (int*)(smem + SB_BYTES + SS_BYTES + TS_BYTES);
    float*          sMin  = (float*)(smem + SB_BYTES + SS_BYTES + 2 * TS_BYTES);
    int*            sMinP = (int*)(smem + SB_BYTES + SS_BYTES + 2 * TS_BYTES + BM * 4);

    int tid = threadIdx.x, lane = tid & 31, wave = tid >> 5;
    int split = blockIdx.x, by = blockIdx.y;

    for (int i = tid; i < BM * TOPK; i += blockDim.x) { sTopS[i] = NEG_INF; sTopI[i] = 0; }
    if (tid < BM) { sMin[tid] = NEG_INF; sMinP[tid] = 0; }

    // Preload A-fragments for this wave's 16-row half (registers, whole N sweep).
    int mhalf = wave >> 2, ctile = wave & 3;
    int mrow = by * BM + mhalf * 16 + (lane & 15);
    const unsigned short* arow = bsb + (size_t)mrow * DK;
    int ha = (lane >= 16) ? 8 : 0;
    VecBF afrag[8];
#pragma unroll
    for (int kk = 0; kk < 8; ++kk) {
        int d0 = kk * 32 + ha;
        afrag[kk].q[0] = *(const uint4*)(arow + d0);
        afrag[kk].q[1] = *(const uint4*)(arow + d0 + 16);
    }

    int colsPerSplit = (N + NSPLIT - 1) / NSPLIT;
    int c0 = split * colsPerSplit;
    int cend = c0 + colsPerSplit; if (cend > N) cend = N;
    int nch = (cend - c0 + BN - 1) / BN;
    int half = (lane >= 16) ? 8 : 0;

#if USE_TDM
    if (tid < 32) {  // wave 0 drives the TDM pipeline (EXEC-independent, per-wave op)
        tdm_stage(idb + (size_t)c0 * DK, (unsigned)(uintptr_t)sB, cend - c0, BN);
    }
#endif
    __syncthreads();

    for (int ci = 0; ci < nch; ++ci) {
        int c = c0 + ci * BN;
        unsigned short* bcur = sB + (ci & 1) * (BN * LDSTR);
#if USE_TDM
        if (tid < 32) __builtin_amdgcn_s_wait_tensorcnt((short)0);
        __syncthreads();
        if ((ci + 1 < nch) && tid < 32) {       // overlap: DMA next chunk during compute
            int cn = c0 + (ci + 1) * BN;
            unsigned short* bnext = sB + ((ci + 1) & 1) * (BN * LDSTR);
            tdm_stage(idb + (size_t)cn * DK, (unsigned)(uintptr_t)bnext, cend - cn, BN);
        }
#else
        for (int i = tid; i < BN * DK / 8; i += blockDim.x) {
            int e = i * 8, row = e / DK, coff = e - row * DK;
            uint4 val = {0, 0, 0, 0};
            if (c + row < cend) val = *(const uint4*)(idb + (size_t)(c + row) * DK + coff);
            *(uint4*)(bcur + row * LDSTR + coff) = val;
        }
        __syncthreads();
#endif
        // ---- WMMA: this wave's 16x16 tile, K=256 in 8 chained steps ----
        const unsigned short* lrow = bcur + (ctile * 16 + (lane & 15)) * LDSTR + half;
        v8f acc = {};
#pragma unroll
        for (int kk = 0; kk < 8; ++kk) {
            VecBF bf;
            bf.q[0] = *(const uint4*)(lrow + kk * 32);      // ds_load_b128
            bf.q[1] = *(const uint4*)(lrow + kk * 32 + 16);
            acc = __builtin_amdgcn_wmma_f32_16x16x32_bf16(false, afrag[kk].v, false, bf.v,
                                                          (short)0, acc, false, false);
        }
        int n = ctile * 16 + (lane & 15);
        int m0 = mhalf * 16 + ((lane >= 16) ? 8 : 0);
        bool valid = (c + n) < cend;
#pragma unroll
        for (int v = 0; v < 8; ++v) sS[(m0 + v) * BN + n] = valid ? acc[v] : NEG_INF;
        __syncthreads();

        // ---- running top-50: one thread per row ----
        // Fast path: branch-free pipelined max over the row; only if it beats the
        // current threshold do the (rare) insertion scan.
        if (tid < BM) {
            int r = tid;
            float mn = sMin[r]; int mp = sMinP[r];
            const float* rowS = sS + r * BN;
            float mx = rowS[0];
#pragma unroll
            for (int j = 1; j < BN; ++j) mx = fmaxf(mx, rowS[j]);
            if (mx > mn) {
                float* tS = sTopS + r * TOPK; int* tI = sTopI + r * TOPK;
                for (int j = 0; j < BN; ++j) {
                    float s = rowS[j];
                    if (s > mn) {
                        tS[mp] = s; tI[mp] = c + j;
                        mn = tS[0]; mp = 0;
                        for (int k = 1; k < TOPK; ++k) {
                            float t = tS[k];
                            if (t < mn) { mn = t; mp = k; }
                        }
                    }
                }
                sMin[r] = mn; sMinP[r] = mp;
            }
        }
        __syncthreads();
    }

    int bbase = by * BM;
    for (int i = tid; i < BM * TOPK; i += blockDim.x) {
        int r = i / TOPK, k = i % TOPK;
        size_t off = ((size_t)(bbase + r) * NSPLIT + split) * TOPK + k;
        pS[off] = sTopS[i];
        pI[off] = sTopI[i];
    }
}

// ---------------- Kernel 4: merge partial top-k, softmax, weighted gather ----
__global__ void merge_kernel(const float* __restrict__ pS, const int* __restrict__ pI,
                             const float* __restrict__ guf, float* __restrict__ out, int N) {
    __shared__ float cs[NSPLIT * TOPK];
    __shared__ int   ci[NSPLIT * TOPK];
    __shared__ float rS[DK];
    __shared__ int   rI[DK];
    __shared__ float sel[TOPK];
    __shared__ int   sIdx[TOPK];
    __shared__ float sP[TOPK];

    int b = blockIdx.x, tid = threadIdx.x;
    for (int i = tid; i < NSPLIT * TOPK; i += blockDim.x) {
        cs[i] = pS[(size_t)b * NSPLIT * TOPK + i];
        ci[i] = pI[(size_t)b * NSPLIT * TOPK + i];
    }
    __syncthreads();

    for (int k = 0; k < TOPK; ++k) {
        float best = NEG_INF; int bp = 0;
        for (int i = tid; i < NSPLIT * TOPK; i += blockDim.x)
            if (cs[i] > best) { best = cs[i]; bp = i; }
        rS[tid] = best; rI[tid] = bp;
        __syncthreads();
        for (int off = DK / 2; off > 0; off >>= 1) {
            if (tid < off && rS[tid + off] > rS[tid]) { rS[tid] = rS[tid + off]; rI[tid] = rI[tid + off]; }
            __syncthreads();
        }
        if (tid == 0) { int p = rI[0]; sel[k] = rS[0]; sIdx[k] = ci[p]; cs[p] = NEG_INF; }
        __syncthreads();
    }

    if (tid == 0) {
        float mx = sel[0], sum = 0.f;
        for (int k = 0; k < TOPK; ++k) { float e = __expf(sel[k] - mx); sP[k] = e; sum += e; }
        float inv = 1.f / sum;
        for (int k = 0; k < TOPK; ++k) sP[k] *= inv;
    }
    __syncthreads();

    if (tid < DK) {
        float acc = 0.f;
        for (int k = 0; k < TOPK; ++k) {
            int r = sIdx[k];
            if ((unsigned)r >= (unsigned)N) r = 0;
            acc += sP[k] * guf[(size_t)r * DK + tid];    // coalesced row gather
        }
        out[(size_t)b * DK + tid] = acc;
    }
}

// ---------------- Host launcher ----------------
extern "C" void kernel_launch(void* const* d_in, const int* in_sizes, int n_in,
                              void* d_out, int out_size, void* d_ws, size_t ws_size,
                              hipStream_t stream) {
    const int*   cur  = (const int*)d_in[0];     // [B,1] int32
    const float* idf  = (const float*)d_in[1];   // [N,256]
    const float* guf  = (const float*)d_in[2];   // [N,256]
    const float* W    = (const float*)d_in[3];   // [256,256]
    const float* bias = (const float*)d_in[4];   // [256]
    int B = in_sizes[0];
    int N = in_sizes[1] / DK;
    float* out = (float*)d_out;

    char* ws = (char*)d_ws;
    size_t o = 0;
    unsigned short* idb = (unsigned short*)(ws + o);
    o += (size_t)N * DK * sizeof(unsigned short); o = (o + 255) & ~(size_t)255;
    unsigned short* bsb = (unsigned short*)(ws + o);
    o += (size_t)B * DK * sizeof(unsigned short); o = (o + 255) & ~(size_t)255;
    float* pS = (float*)(ws + o);
    o += (size_t)B * NSPLIT * TOPK * sizeof(float); o = (o + 255) & ~(size_t)255;
    int* pI = (int*)(ws + o);

    long long n8 = ((long long)N * DK) / 8;
    cvt_kernel<<<dim3(4096), dim3(256), 0, stream>>>(idf, idb, n8);
    bs_kernel<<<dim3(DK / 16, B / 16), dim3(32), 0, stream>>>(cur, idf, W, bias, bsb, B, N);
    score_topk_kernel<<<dim3(NSPLIT, B / BM), dim3(256), SMEM_TOTAL, stream>>>(bsb, idb, pS, pI, N);
    merge_kernel<<<dim3(B), dim3(DK), 0, stream>>>(pS, pI, guf, out, N);
}
